// simpleLSTM_11879879544202
// MI455X (gfx1250) — matrix operand order repaired and checked by reading, not compiled, not run on hardware
//
#include <hip/hip_runtime.h>
#include <hip/hip_bf16.h>

// ---------------- problem dims ----------------
#define BQ     256
#define SRC_T  512
#define IN_D   64
#define HID    512
#define TGT_TT 96
#define G4H    2048   // 4*HID

typedef __attribute__((ext_vector_type(16))) _Float16 v16h;
typedef __attribute__((ext_vector_type(8)))  _Float16 v8h;
typedef __attribute__((ext_vector_type(8)))  float    v8f;

// A fragment (16x32 f16, contiguous k order, same as B): lane L (m=L&15, khalf=L>>4):
// elems 0..15 = A[m, k0+khalf*16 .. +15]
__device__ __forceinline__ v16h load_a_frag(const _Float16* p) {
    v8h lo = *(const v8h*)p;
    v8h hi = *(const v8h*)(p + 8);
    v16h a;
#pragma unroll
    for (int i = 0; i < 8; ++i) { a[i] = lo[i]; a[8 + i] = hi[i]; }
    return a;
}

__device__ __forceinline__ float sigm(float x) { return 1.0f / (1.0f + __expf(-x)); }

// ================= fused LSTM step: GEMM (WMMA) + cell update =================
// gates[b, n + 512*j] = sum_k A[b,k] * W[n + 512*j, k]   (j = i,f,g,o)
// A operand split at KSPLIT (folds at compile time inside the unrolled loop).
// Wave tile: 16 rows x 16 hidden cols; the wave's 4 WMMA accumulators map to
// the 4 gate column-groups => each lane register ends holding (gi,gf,gg,go)
// for one (b, n): cell update in-register, no gates buffer.
// K loop fully unrolled + manually double-buffered; sched_barriers pin the
// issue order  loads(i+1) | wmma(i) | loads(i+2) | ...  so one full fragment
// group stays outstanding across every wmma group (in-wave latency hiding).
// Grid: (HID/32, BQ/64), 256 threads (8 waves: 4 along M x 2 along n).
template <int KSPLIT, int KTOT, int LDA1, int LDA2,
          bool HAS_X, bool HAS_D1, bool HAS_RELU>
__global__ __launch_bounds__(256) void k_lstm_step(
    const _Float16* __restrict__ A1,
    const _Float16* __restrict__ A2,
    const _Float16* __restrict__ W,
    const float* __restrict__ bih, const float* __restrict__ bhh,
    const float* __restrict__ xvec, const float* __restrict__ wih_col, int xstride,
    float* __restrict__ c,
    _Float16* __restrict__ hdst0, int ld0, int off0,
    _Float16* __restrict__ hdst1, int ld1, int off1,
    _Float16* __restrict__ hrelu)
{
    const int lane = threadIdx.x & 31;
    const int wave = threadIdx.x >> 5;
    const int wm = wave & 3, wn = wave >> 2;
    const int m0 = blockIdx.y * 64 + wm * 16;
    const int n0 = blockIdx.x * 32 + wn * 16;
    const int mrow  = m0 + (lane & 15);
    const int khalf = lane >> 4;
    const int ncol  = n0 + (lane & 15);   // hidden-unit column

    const _Float16* a1p = A1 + (size_t)mrow * LDA1 + khalf * 16;
    const _Float16* a2p = A2 + (size_t)mrow * LDA2 + khalf * 16;
    const _Float16* wp  = W  + (size_t)ncol * KTOT + khalf * 16;

    auto aptr = [&](int k0) {
        return (k0 < KSPLIT) ? (a1p + k0) : (a2p + (k0 - KSPLIT));
    };
    auto bptr = [&](int j, int k0) {
        return (const v16h*)(wp + (size_t)j * HID * KTOT + k0);
    };

    v8f acc[4];
#pragma unroll
    for (int j = 0; j < 4; ++j)
#pragma unroll
        for (int r = 0; r < 8; ++r) acc[j][r] = 0.0f;

    constexpr int NK = KTOT / 32;
    v16h a_cur = load_a_frag(aptr(0));
    v16h b_cur[4];
#pragma unroll
    for (int j = 0; j < 4; ++j) b_cur[j] = *bptr(j, 0);

#pragma unroll
    for (int it = 0; it < NK; ++it) {
        v16h a_nxt, b_nxt[4];
        if (it + 1 < NK) {                       // compile-time folded
            const int kn = (it + 1) * 32;
            a_nxt = load_a_frag(aptr(kn));
#pragma unroll
            for (int j = 0; j < 4; ++j) b_nxt[j] = *bptr(j, kn);
        }
        __builtin_amdgcn_sched_barrier(0);       // loads(i+1) stay above wmma(i)
#pragma unroll
        for (int j = 0; j < 4; ++j)
            acc[j] = __builtin_amdgcn_wmma_f32_16x16x32_f16(
                false, a_cur, false, b_cur[j], (short)0, acc[j], false, false);
        __builtin_amdgcn_sched_barrier(0);       // loads(i+2) stay below wmma(i)
        if (it + 1 < NK) {
            a_cur = a_nxt;
#pragma unroll
            for (int j = 0; j < 4; ++j) b_cur[j] = b_nxt[j];
        }
    }

    // D layout: lane l -> col = l&15 ; row = m0 + (l>>4)*8 + r
    const int mbase = m0 + (lane >> 4) * 8;
    const int n = ncol;
    const float bi = bih[n]            + bhh[n];
    const float bf = bih[n + HID]      + bhh[n + HID];
    const float bg = bih[n + 2 * HID]  + bhh[n + 2 * HID];
    const float bo = bih[n + 3 * HID]  + bhh[n + 3 * HID];
    float wci = 0.f, wcf = 0.f, wcg = 0.f, wco = 0.f;
    if constexpr (HAS_X) {
        wci = wih_col[n];           wcf = wih_col[n + HID];
        wcg = wih_col[n + 2 * HID]; wco = wih_col[n + 3 * HID];
    }
#pragma unroll
    for (int r = 0; r < 8; ++r) {
        const int b = mbase + r;                  // batch row
        float gi = acc[0][r] + bi;
        float gf = acc[1][r] + bf;
        float gg = acc[2][r] + bg;
        float go = acc[3][r] + bo;
        if constexpr (HAS_X) {
            float x = xvec[(size_t)b * xstride];
            gi += x * wci; gf += x * wcf; gg += x * wcg; go += x * wco;
        }
        float i = sigm(gi);
        float f = sigm(gf);
        float g = tanhf(gg);
        float o = sigm(go);
        float cn = f * c[(size_t)b * HID + n] + i * g;
        c[(size_t)b * HID + n] = cn;
        float h = o * tanhf(cn);
        hdst0[(size_t)b * ld0 + off0 + n] = (_Float16)h;
        if constexpr (HAS_D1) hdst1[(size_t)b * ld1 + off1 + n] = (_Float16)h;
        if constexpr (HAS_RELU) hrelu[(size_t)b * HID + n] = (_Float16)(h > 0.0f ? h : 0.0f);
    }
}

// ============== fc0: Cf16[256,512] = relu(A[256,512] @ W[512,512]^T + b) =====
// Grid: (512/128, 256/64), 256 threads (8 waves: 4 along M x 2 along N).
__global__ __launch_bounds__(256) void k_fc0_gemm(
    const _Float16* __restrict__ A, const _Float16* __restrict__ W,
    _Float16* __restrict__ Cf16, const float* __restrict__ bias)
{
    const int lane = threadIdx.x & 31;
    const int wave = threadIdx.x >> 5;
    const int wm = wave & 3, wn = wave >> 2;
    const int m0 = blockIdx.y * 64 + wm * 16;
    const int n0 = blockIdx.x * 128 + wn * 64;
    const int mrow  = m0 + (lane & 15);
    const int khalf = lane >> 4;

    const _Float16* ap = A + (size_t)mrow * HID + khalf * 16;
    const _Float16* wp = W + (size_t)(n0 + (lane & 15)) * HID + khalf * 16;

    auto bptr = [&](int j, int k0) {
        return (const v16h*)(wp + (size_t)j * 16 * HID + k0);
    };

    v8f acc[4];
#pragma unroll
    for (int j = 0; j < 4; ++j)
#pragma unroll
        for (int r = 0; r < 8; ++r) acc[j][r] = 0.0f;

    constexpr int NK = HID / 32;
    v16h a_cur = load_a_frag(ap);
    v16h b_cur[4];
#pragma unroll
    for (int j = 0; j < 4; ++j) b_cur[j] = *bptr(j, 0);

#pragma unroll
    for (int it = 0; it < NK; ++it) {
        v16h a_nxt, b_nxt[4];
        if (it + 1 < NK) {
            const int kn = (it + 1) * 32;
            a_nxt = load_a_frag(ap + kn);
#pragma unroll
            for (int j = 0; j < 4; ++j) b_nxt[j] = *bptr(j, kn);
        }
        __builtin_amdgcn_sched_barrier(0);
#pragma unroll
        for (int j = 0; j < 4; ++j)
            acc[j] = __builtin_amdgcn_wmma_f32_16x16x32_f16(
                false, a_cur, false, b_cur[j], (short)0, acc[j], false, false);
        __builtin_amdgcn_sched_barrier(0);
        if (it + 1 < NK) {
            a_cur = a_nxt;
#pragma unroll
            for (int j = 0; j < 4; ++j) b_cur[j] = b_nxt[j];
        }
    }

    const int mbase = m0 + (lane >> 4) * 8;
#pragma unroll
    for (int j = 0; j < 4; ++j) {
        const int nc = n0 + j * 16 + (lane & 15);
        const float bb = bias[nc];
#pragma unroll
        for (int r = 0; r < 8; ++r) {
            float v = acc[j][r] + bb;
            v = v > 0.0f ? v : 0.0f;
            Cf16[(size_t)(mbase + r) * HID + nc] = (_Float16)v;
        }
    }
}

// ---------------- small helpers ----------------
__global__ void k_cvt(const float* __restrict__ s, _Float16* __restrict__ d,
                      int rows, int cols, int dld, int doff) {
    int idx = blockIdx.x * blockDim.x + threadIdx.x;
    if (idx >= rows * cols) return;
    int r = idx / cols, c = idx % cols;
    d[(size_t)r * dld + doff + c] = (_Float16)s[idx];
}
__global__ void k_cvt_flat(const float* __restrict__ s, _Float16* __restrict__ d, int n) {
    int i = blockIdx.x * blockDim.x + threadIdx.x;
    if (i < n) d[i] = (_Float16)s[i];
}
__global__ void k_zero_f32(float* p, int n) {
    int i = blockIdx.x * blockDim.x + threadIdx.x;
    if (i < n) p[i] = 0.0f;
}
__global__ void k_zero_f16(_Float16* p, int n) {
    int i = blockIdx.x * blockDim.x + threadIdx.x;
    if (i < n) p[i] = (_Float16)0.0f;
}

// fc1: y[b] = dot(fcB[b,:], W1) + b1 ; writes out[:,t] and ybuf (next-step x)
__global__ __launch_bounds__(256) void k_fc1(
    const _Float16* __restrict__ fcB, const float* __restrict__ W1,
    const float* __restrict__ b1, float* __restrict__ out,
    float* __restrict__ ybuf, int t)
{
    int b = threadIdx.x;   // 256 threads, 1 block
    float s = b1[0];
    for (int k = 0; k < HID; ++k) s += (float)fcB[(size_t)b * HID + k] * W1[k];
    out[(size_t)b * TGT_TT + t] = s;
    ybuf[b] = s;
}

// ---------------- host ----------------
extern "C" void kernel_launch(void* const* d_in, const int* in_sizes, int n_in,
                              void* d_out, int out_size, void* d_ws, size_t ws_size,
                              hipStream_t stream) {
    const float* src   = (const float*)d_in[0];
    const float* tgt   = (const float*)d_in[1];
    const float* eWih0 = (const float*)d_in[2];
    const float* eWhh0 = (const float*)d_in[3];
    const float* ebih0 = (const float*)d_in[4];
    const float* ebhh0 = (const float*)d_in[5];
    const float* dWih0 = (const float*)d_in[6];   // [2048,1]
    const float* dWhh0 = (const float*)d_in[7];
    const float* dbih0 = (const float*)d_in[8];
    const float* dbhh0 = (const float*)d_in[9];
    const float* eWih1 = (const float*)d_in[10];
    const float* eWhh1 = (const float*)d_in[11];
    const float* ebih1 = (const float*)d_in[12];
    const float* ebhh1 = (const float*)d_in[13];
    const float* dWih1 = (const float*)d_in[14];
    const float* dWhh1 = (const float*)d_in[15];
    const float* dbih1 = (const float*)d_in[16];
    const float* dbhh1 = (const float*)d_in[17];
    const float* fcW0  = (const float*)d_in[18];
    const float* fcb0  = (const float*)d_in[19];
    const float* fcW1  = (const float*)d_in[20];
    const float* fcb1  = (const float*)d_in[21];
    float* out = (float*)d_out;

    // workspace carve (256B aligned)
    char* wsp = (char*)d_ws;
    auto carve = [&](size_t bytes) -> char* {
        char* p = wsp;
        wsp += (bytes + 255) & ~(size_t)255;
        return p;
    };
    _Float16* Wcat0e = (_Float16*)carve((size_t)G4H * (IN_D + HID) * 2); // [2048,576]
    _Float16* Wcat1e = (_Float16*)carve((size_t)G4H * (2 * HID) * 2);    // [2048,1024]
    _Float16* Wd0    = (_Float16*)carve((size_t)G4H * HID * 2);          // [2048,512]
    _Float16* Wcat1d = (_Float16*)carve((size_t)G4H * (2 * HID) * 2);    // [2048,1024]
    _Float16* Wfc0h  = (_Float16*)carve((size_t)HID * HID * 2);          // [512,512]
    _Float16* srcH   = (_Float16*)carve((size_t)BQ * SRC_T * IN_D * 2);  // f16 src
    _Float16* Ah0e   = (_Float16*)carve((size_t)BQ * HID * 2);           // h0 [256,512]
    _Float16* A1e    = (_Float16*)carve((size_t)BQ * (2 * HID) * 2);     // [h0|h1] [256,1024]
    _Float16* Afc    = (_Float16*)carve((size_t)BQ * HID * 2);           // relu(h1)
    _Float16* fcB    = (_Float16*)carve((size_t)BQ * HID * 2);           // fc0 out
    float*    c0     = (float*)carve((size_t)BQ * HID * 4);
    float*    c1     = (float*)carve((size_t)BQ * HID * 4);
    float*    ybuf   = (float*)carve((size_t)BQ * 4);

    const int T = 256;
    auto blocks = [](int n, int t) { return (n + t - 1) / t; };

    // ---- one-time: weights + src to f16 ----
    k_cvt<<<blocks(G4H * IN_D, T), T, 0, stream>>>(eWih0, Wcat0e, G4H, IN_D, IN_D + HID, 0);
    k_cvt<<<blocks(G4H * HID, T), T, 0, stream>>>(eWhh0, Wcat0e, G4H, HID, IN_D + HID, IN_D);
    k_cvt<<<blocks(G4H * HID, T), T, 0, stream>>>(eWih1, Wcat1e, G4H, HID, 2 * HID, 0);
    k_cvt<<<blocks(G4H * HID, T), T, 0, stream>>>(eWhh1, Wcat1e, G4H, HID, 2 * HID, HID);
    k_cvt<<<blocks(G4H * HID, T), T, 0, stream>>>(dWhh0, Wd0, G4H, HID, HID, 0);
    k_cvt<<<blocks(G4H * HID, T), T, 0, stream>>>(dWih1, Wcat1d, G4H, HID, 2 * HID, 0);
    k_cvt<<<blocks(G4H * HID, T), T, 0, stream>>>(dWhh1, Wcat1d, G4H, HID, 2 * HID, HID);
    k_cvt<<<blocks(HID * HID, T), T, 0, stream>>>(fcW0, Wfc0h, HID, HID, HID, 0);
    k_cvt_flat<<<blocks(BQ * SRC_T * IN_D, T), T, 0, stream>>>(src, srcH, BQ * SRC_T * IN_D);

    // ---- zero-init state ----
    k_zero_f32<<<blocks(BQ * HID, T), T, 0, stream>>>(c0, BQ * HID);
    k_zero_f32<<<blocks(BQ * HID, T), T, 0, stream>>>(c1, BQ * HID);
    k_zero_f16<<<blocks(BQ * HID, T), T, 0, stream>>>(Ah0e, BQ * HID);
    k_zero_f16<<<blocks(BQ * 2 * HID, T), T, 0, stream>>>(A1e, BQ * 2 * HID);

    const dim3 gLstm(HID / 32, BQ / 64);   // (16,4) x 256 thr
    const dim3 gFc0(HID / 128, BQ / 64);   // (4,4)  x 256 thr

    // ---- encoder: 2 fused kernels per timestep ----
    for (int t = 0; t < SRC_T; ++t) {
        // layer 0: A = [src[:,t,:] | h0] (split-K), W = [Wih0|Whh0], K=576
        k_lstm_step<IN_D, IN_D + HID, SRC_T * IN_D, HID, false, true, false>
            <<<gLstm, T, 0, stream>>>(
            srcH + (size_t)t * IN_D,                 // k<64: src row
            Ah0e,                                    // k>=64: h0
            Wcat0e,
            ebih0, ebhh0, nullptr, nullptr, 0, c0,
            Ah0e, HID, 0,                            // h0 -> layer0 A (next step)
            A1e, 2 * HID, 0,                         // h0 -> layer1 A (this step)
            nullptr);
        // layer 1: A = [h0|h1], W = [Wih1|Whh1], K=1024
        k_lstm_step<2 * HID, 2 * HID, 2 * HID, 2 * HID, false, false, false>
            <<<gLstm, T, 0, stream>>>(
            A1e, A1e, Wcat1e,
            ebih1, ebhh1, nullptr, nullptr, 0, c1,
            A1e, 2 * HID, HID,                       // h1 -> layer1 A (next step)
            nullptr, 0, 0, nullptr);
    }

    // ---- autoregressive decoder (reuses Ah0e / A1e: they hold enc final h) ----
    for (int t = 0; t < TGT_TT; ++t) {
        const float* xp = (t == 0) ? tgt : ybuf;
        const int xs = (t == 0) ? TGT_TT : 1;
        // layer 0: gates = h0 @ Whh0^T ; x*Wih0[:,0] added in epilogue
        k_lstm_step<HID, HID, HID, HID, true, true, false>
            <<<gLstm, T, 0, stream>>>(
            Ah0e, Ah0e, Wd0,
            dbih0, dbhh0, xp, dWih0, xs, c0,
            Ah0e, HID, 0,                            // h0 -> layer0 A (next step)
            A1e, 2 * HID, 0,                         // h0 -> layer1 A (this step)
            nullptr);
        // layer 1: A = [h0|h1], W = [dWih1|dWhh1], K=1024
        k_lstm_step<2 * HID, 2 * HID, 2 * HID, 2 * HID, false, false, true>
            <<<gLstm, T, 0, stream>>>(
            A1e, A1e, Wcat1d,
            dbih1, dbhh1, nullptr, nullptr, 0, c1,
            A1e, 2 * HID, HID,                       // h1 (raw) -> layer1 A
            nullptr, 0, 0,
            Afc);                                    // relu(h1) -> fc input
        // fc head
        k_fc0_gemm<<<gFc0, T, 0, stream>>>(Afc, Wfc0h, fcB, fcb0);
        k_fc1<<<1, 256, 0, stream>>>(fcB, fcW1, fcb1, out, ybuf, t);
    }
}
